// CrystalHypergraphConv_17291538334096
// MI455X (gfx1250) — matrix operand order repaired and checked by reading, not compiled, not run on hardware
//
#include <hip/hip_runtime.h>
#include <hip/hip_bf16.h>

// CrystalHypergraphConv forward for MI455X (gfx1250, wave32, WMMA).
// All GEMMs via v_wmma_f32_16x16x32_f16 with f32 accumulation; f16 operand
// copies halve the random-gather traffic (the true bottleneck vs 23.3 TB/s).
// Hot GEMMs double-buffer the gathers in registers so global latency overlaps
// the WMMA chain; global_prefetch_b8 warms the index/msg streams.

#define NN    100000   // nodes
#define NH    200000   // hyperedges
#define NE    1000000  // incidences
#define NG    256      // graphs
#define HD    64       // h_dim
#define HEF   35       // hedge feature dim
#define HOUTD 128
#define NLAY  3
#define EPSV  1e-5f

typedef __attribute__((ext_vector_type(16))) _Float16 v16h;
typedef __attribute__((ext_vector_type(8)))  float    v8f;
typedef _Float16 h16;

__device__ __forceinline__ float sigmoid_(float x) { return 1.0f / (1.0f + __expf(-x)); }
__device__ __forceinline__ float softplus_(float x) { return (x > 20.f) ? x : log1pf(__expf(x)); }

// ---- WMMA fragment builders (contiguous k = 16*(lane/16) + e in both A and B, wave32) ----
// A 16x32 f16: lane L holds row m=L&15; lanes<16 carry K 0..15,
// lanes>=16 carry K 16..31; pairs packed per VGPR.
__device__ __forceinline__ v16h frag_a_lds(const h16* s, int rowStride, int step, int lane) {
  const int m = lane & 15, g = lane >> 4;
  const h16* row = s + m * rowStride + step * 32;
  v16h a;
#pragma unroll
  for (int e = 0; e < 16; ++e) {
    const int v = e >> 1, p = e & 1;
    const int lk = 16 * g + 2 * v + p;
    a[e] = row[lk];
  }
  return a;
}
// B 32x16 f16: lane L holds column n=L&15; lanes<16 K=0..15, lanes>=16 K=16..31.
__device__ __forceinline__ v16h frag_b_glb(const h16* __restrict__ W, int ldn, int step, int n0, int lane) {
  const int n = n0 + (lane & 15), g = lane >> 4;
  v16h b;
#pragma unroll
  for (int e = 0; e < 16; ++e) {
    const int k = step * 32 + 16 * g + e;
    b[e] = W[(size_t)k * ldn + n];
  }
  return b;
}
#define WMMA32(A, B, C) __builtin_amdgcn_wmma_f32_16x16x32_f16(false, (A), false, (B), (short)0, (C), false, false)
#define LDS_FENCE() asm volatile("s_wait_dscnt 0x0" ::: "memory")

// ---------------- utility kernels ----------------
__global__ void k_convert16(const float* __restrict__ src, h16* __restrict__ dst,
                            int K, int N, int Kp, int Np) {
  int i = blockIdx.x * blockDim.x + threadIdx.x;
  if (i >= Kp * Np) return;
  int k = i / Np, n = i % Np;
  dst[i] = (k < K && n < N) ? (h16)src[k * N + n] : (h16)0.f;
}

__global__ void k_inc_counts(const int* __restrict__ ni, const int* __restrict__ hi,
                             float* __restrict__ ncnt, float* __restrict__ hcnt, int n) {
  int i = blockIdx.x * blockDim.x + threadIdx.x;
  if (i >= n) return;
  atomicAdd(&ncnt[ni[i]], 1.f);
  atomicAdd(&hcnt[hi[i]], 1.f);
}

__global__ void k_batch_counts(const int* __restrict__ b, float* __restrict__ pcnt, int n) {
  int i = blockIdx.x * blockDim.x + threadIdx.x;
  if (i < n) atomicAdd(&pcnt[b[i]], 1.f);
}

// per-column mean / rstd (training-mode BN, biased variance)
__global__ void k_colstats(const float* __restrict__ z, int M, int C,
                           float* __restrict__ mean, float* __restrict__ rstd) {
  int c = blockIdx.x;
  float s = 0.f, s2 = 0.f;
  for (int r = threadIdx.x; r < M; r += blockDim.x) {
    float v = z[(size_t)r * C + c];
    s += v; s2 += v * v;
  }
  __shared__ float a1[256], a2[256];
  a1[threadIdx.x] = s; a2[threadIdx.x] = s2;
  __syncthreads();
  for (int w = 128; w > 0; w >>= 1) {
    if ((int)threadIdx.x < w) { a1[threadIdx.x] += a1[threadIdx.x + w]; a2[threadIdx.x] += a2[threadIdx.x + w]; }
    __syncthreads();
  }
  if (threadIdx.x == 0) {
    float m = a1[0] / (float)M;
    float v = a2[0] / (float)M - m * m;
    mean[c] = m; rstd[c] = rsqrtf(v + EPSV);
  }
}

// ---------------- embed: x[NN,92] @ W[92,64] + b (WMMA, Kp=96) ----------------
__global__ void k_embed(const float* __restrict__ x, const h16* __restrict__ wemb,
                        const float* __restrict__ eb, float* __restrict__ xc,
                        h16* __restrict__ xh, int totalWaves) {
  __shared__ __align__(16) h16 sA[4][16 * 96];
  const int wv = threadIdx.x >> 5, lane = threadIdx.x & 31;
  const int w = blockIdx.x * 4 + wv;
  const int ntile = w & 3, sub = w >> 2, nsub = totalWaves >> 2;
  const int g = lane >> 4, col = ntile * 16 + (lane & 15);
  v16h B[3];
#pragma unroll
  for (int s = 0; s < 3; ++s) B[s] = frag_b_glb(wemb, 64, s, ntile * 16, lane);
  const float ebc = eb[col];
  for (int tile = sub; tile < NN / 16; tile += nsub) {
    {
      const int r = lane & 15, hh = lane >> 4;
      const int row = tile * 16 + r;
#pragma unroll
      for (int c = 0; c < 48; ++c) {
        const int gc = hh * 48 + c;
        float v = (gc < 92) ? x[(size_t)row * 92 + gc] : 0.f;
        sA[wv][r * 96 + gc] = (h16)v;
      }
    }
    LDS_FENCE();
    v8f acc = {};
#pragma unroll
    for (int s = 0; s < 3; ++s) {
      v16h A = frag_a_lds(&sA[wv][0], 96, s, lane);
      acc = WMMA32(A, B[s], acc);
    }
#pragma unroll
    for (int rr = 0; rr < 8; ++rr) {
      const int row = tile * 16 + rr + 8 * g;
      const float v = acc[rr] + ebc;
      xc[(size_t)row * HD + col] = v;
      xh[(size_t)row * HD + col] = (h16)v;
    }
  }
}

// ---------------- hedge side: gather-mean, msg build ----------------
__global__ void k_hedge_gather(const int* __restrict__ ni, const int* __restrict__ hi,
                               const float* __restrict__ xc, float* __restrict__ hsum, int n) {
  int i = blockIdx.x * blockDim.x + threadIdx.x;
  if (i >= n) return;
  const int nd = ni[i], hd = hi[i];
  const float4* src = (const float4*)(xc + (size_t)nd * HD);
  float* dst = hsum + (size_t)hd * HD;
#pragma unroll
  for (int q = 0; q < 16; ++q) {
    float4 v = src[q];
    atomicAdd(&dst[q * 4 + 0], v.x);
    atomicAdd(&dst[q * 4 + 1], v.y);
    atomicAdd(&dst[q * 4 + 2], v.z);
    atomicAdd(&dst[q * 4 + 3], v.w);
  }
}

// msg[h][0..63]=mean, [64..98]=hedge_attr, [99..127]=0  (f16, K padded to 128)
__global__ void k_hedge_msg(const float* __restrict__ hsum, const float* __restrict__ hcnt,
                            const float* __restrict__ hec, h16* __restrict__ msg, int n) {
  int i = blockIdx.x * blockDim.x + threadIdx.x;
  if (i >= n) return;
  const int h = i >> 7, c = i & 127;
  float v;
  if (c < HD)            v = hsum[(size_t)h * HD + c] / fmaxf(hcnt[h], 1.f);
  else if (c < HD + HEF) v = hec[(size_t)h * HEF + (c - HD)];
  else                   v = 0.f;
  msg[i] = (h16)v;
}

// ---------------- hedge GEMM: msg[NH,128] @ {f1,c1}[128,48] (valid 99x35) ----
__global__ void k_hedge_gemm(const h16* __restrict__ msg, const h16* __restrict__ wf,
                             const h16* __restrict__ wc, float* __restrict__ zf,
                             float* __restrict__ zc, int totalWaves) {
  __shared__ __align__(16) h16 sA[4][16 * 128];
  const int wv = threadIdx.x >> 5, lane = threadIdx.x & 31;
  const int w = blockIdx.x * 4 + wv;
  const int ntile = w % 3, sub = w / 3, nsub = totalWaves / 3;
  const int g = lane >> 4, col = ntile * 16 + (lane & 15);
  const int r = lane & 15, hh = lane >> 4;
  v16h BF[4], BC[4];
#pragma unroll
  for (int s = 0; s < 4; ++s) {
    BF[s] = frag_b_glb(wf, 48, s, ntile * 16, lane);
    BC[s] = frag_b_glb(wc, 48, s, ntile * 16, lane);
  }
  const int NT = NH / 16;
  auto fetch = [&](int tile, uint4* b) {
    const uint4* src = (const uint4*)(msg + (size_t)(tile * 16 + r) * 128 + hh * 64);
#pragma unroll
    for (int q = 0; q < 8; ++q) b[q] = src[q];
  };
  if (sub >= NT) return;
  uint4 cur[8];
  fetch(sub, cur);
  for (int tile = sub; tile < NT; tile += nsub) {
    {
      uint4* dst = (uint4*)(&sA[wv][r * 128 + hh * 64]);
#pragma unroll
      for (int q = 0; q < 8; ++q) dst[q] = cur[q];
    }
    LDS_FENCE();
    // overlap next tile's gather with this tile's WMMA chain
    const int tn = tile + nsub;
    const bool hasNext = tn < NT;
    uint4 nxt[8];
    if (hasNext) {
      fetch(tn, nxt);
      const int t2 = tn + nsub;
      if (t2 < NT)
        __builtin_prefetch(msg + (size_t)(t2 * 16 + r) * 128 + hh * 64, 0, 0);
    }
    v8f aF = {}, aC = {};
#pragma unroll
    for (int s = 0; s < 4; ++s) {
      v16h A = frag_a_lds(&sA[wv][0], 128, s, lane);
      aF = WMMA32(A, BF[s], aF);
      aC = WMMA32(A, BC[s], aC);
    }
    if (col < HEF) {
#pragma unroll
      for (int rr = 0; rr < 8; ++rr) {
        const int row = tile * 16 + rr + 8 * g;
        zf[(size_t)row * HEF + col] = aF[rr];
        zc[(size_t)row * HEF + col] = aC[rr];
      }
    }
    if (hasNext) {
#pragma unroll
      for (int q = 0; q < 8; ++q) cur[q] = nxt[q];
    }
  }
}

// BN + gate: he = sigmoid(bn(zf)) * softplus(bn(zc)); f1_b/c1_b absorbed by BN
__global__ void k_hedge_act(const float* __restrict__ zf, const float* __restrict__ zc,
                            const float* __restrict__ mf, const float* __restrict__ rf,
                            const float* __restrict__ mc, const float* __restrict__ rc,
                            const float* __restrict__ gF, const float* __restrict__ bF,
                            const float* __restrict__ gC, const float* __restrict__ bC,
                            float* __restrict__ hec, h16* __restrict__ heh, int n) {
  int i = blockIdx.x * blockDim.x + threadIdx.x;
  if (i >= n) return;
  const int h = i / HEF, c = i % HEF;
  float f = (zf[i] - mf[c]) * rf[c] * gF[c] + bF[c];
  float s = (zc[i] - mc[c]) * rc[c] * gC[c] + bC[c];
  float v = sigmoid_(f) * softplus_(s);
  hec[i] = v;
  heh[(size_t)h * 48 + c] = (h16)v;   // padded row (cols 35..47 stay 0)
}

// ---- edge GEMM fused with gather + gate + scatter-add into node sums ----
// z=[x_half[node] | he_half[hedge] | 0] (K=128);  o = sig(z@f2+b)*sp(z@c2+b)
__global__ void k_edge_gemm(const int* __restrict__ ni, const int* __restrict__ hi,
                            const h16* __restrict__ xh, const h16* __restrict__ heh,
                            const h16* __restrict__ wf, const h16* __restrict__ wc,
                            const float* __restrict__ bf, const float* __restrict__ bc,
                            float* __restrict__ nsum, int totalWaves) {
  __shared__ __align__(16) h16 sA[4][16 * 128];
  __shared__ int sIdx[4][16];
  const int wv = threadIdx.x >> 5, lane = threadIdx.x & 31;
  const int w = blockIdx.x * 4 + wv;
  const int ntile = w & 3, sub = w >> 2, nsub = totalWaves >> 2;
  const int g = lane >> 4, col = ntile * 16 + (lane & 15);
  const int r = lane & 15, hh = lane >> 4;
  v16h BF[4], BC[4];
#pragma unroll
  for (int s = 0; s < 4; ++s) {
    BF[s] = frag_b_glb(wf, 64, s, ntile * 16, lane);
    BC[s] = frag_b_glb(wc, 64, s, ntile * 16, lane);
  }
  const float bfc = bf[col], bcc = bc[col];
  const int NT = NE / 16;
  auto fetch = [&](int tile, uint4* b, int& ndOut) {
    const int e = tile * 16 + r;
    if (hh == 0) {                       // lanes 0..15: node feature half
      const int nd = ni[e];
      ndOut = nd;
      const uint4* src = (const uint4*)(xh + (size_t)nd * HD);
#pragma unroll
      for (int q = 0; q < 8; ++q) b[q] = src[q];
    } else {                             // lanes 16..31: hedge feature half (+pad)
      const int hd = hi[e];
      ndOut = 0;
      const uint4* src = (const uint4*)(heh + (size_t)hd * 48);
#pragma unroll
      for (int q = 0; q < 6; ++q) b[q] = src[q];
      b[6] = make_uint4(0u, 0u, 0u, 0u);
      b[7] = b[6];
    }
  };
  if (sub >= NT) return;
  uint4 cur[8];
  int ndCur;
  fetch(sub, cur, ndCur);
  for (int tile = sub; tile < NT; tile += nsub) {
    {
      if (hh == 0) sIdx[wv][r] = ndCur;
      uint4* dst = (uint4*)(&sA[wv][r * 128 + hh * 64]);
#pragma unroll
      for (int q = 0; q < 8; ++q) dst[q] = cur[q];
    }
    LDS_FENCE();
    // overlap next tile's random-row gather with the WMMA chain
    const int tn = tile + nsub;
    const bool hasNext = tn < NT;
    uint4 nxt[8];
    int ndNext = 0;
    if (hasNext) {
      fetch(tn, nxt, ndNext);
      const int t2 = tn + nsub;
      if (t2 < NT) {
        __builtin_prefetch(&ni[t2 * 16 + r], 0, 0);
        __builtin_prefetch(&hi[t2 * 16 + r], 0, 0);
      }
    }
    v8f aF = {}, aC = {};
#pragma unroll
    for (int s = 0; s < 4; ++s) {
      v16h A = frag_a_lds(&sA[wv][0], 128, s, lane);
      aF = WMMA32(A, BF[s], aF);
      aC = WMMA32(A, BC[s], aC);
    }
#pragma unroll
    for (int rr = 0; rr < 8; ++rr) {
      const int m = rr + 8 * g;
      const int nd = sIdx[wv][m];
      const float o = sigmoid_(aF[rr] + bfc) * softplus_(aC[rr] + bcc);
      atomicAdd(&nsum[(size_t)nd * HD + col], o);
    }
    if (hasNext) {
      ndCur = ndNext;
#pragma unroll
      for (int q = 0; q < 8; ++q) cur[q] = nxt[q];
    }
  }
}

__global__ void k_node_finalize(float* __restrict__ nsum, const float* __restrict__ ncnt, int n) {
  int i = blockIdx.x * blockDim.x + threadIdx.x;
  if (i < n) nsum[i] /= fmaxf(ncnt[i >> 6], 1.f);
}

// x = softplus(bn(o) + x)
__global__ void k_node_act(const float* __restrict__ o, const float* __restrict__ mo,
                           const float* __restrict__ ro, const float* __restrict__ gO,
                           const float* __restrict__ bO, float* __restrict__ xc,
                           h16* __restrict__ xh, int n) {
  int i = blockIdx.x * blockDim.x + threadIdx.x;
  if (i >= n) return;
  const int c = i & 63;
  float ob = (o[i] - mo[c]) * ro[c] * gO[c] + bO[c];
  float xn = softplus_(ob + xc[i]);
  xc[i] = xn;
  xh[i] = (h16)xn;
}

__global__ void k_pool(const int* __restrict__ batch, const float* __restrict__ xc,
                       float* __restrict__ psum, int n) {
  int i = blockIdx.x * blockDim.x + threadIdx.x;
  if (i >= n) return;
  atomicAdd(&psum[(size_t)batch[i >> 6] * HD + (i & 63)], xc[i]);
}

// head: pooled mean -> sp(@l2_w+b) -> @out_w+b ; one block per graph
__global__ void k_head(const float* __restrict__ psum, const float* __restrict__ pcnt,
                       const float* __restrict__ l2w, const float* __restrict__ l2b,
                       const float* __restrict__ ow, const float* __restrict__ ob,
                       float* __restrict__ out) {
  const int gidx = blockIdx.x, j = threadIdx.x;  // 128 threads
  __shared__ float pl[HD];
  __shared__ float red[HOUTD];
  if (j < HD) pl[j] = psum[(size_t)gidx * HD + j] / fmaxf(pcnt[gidx], 1.f);
  __syncthreads();
  float d = l2b[j];
  for (int c = 0; c < HD; ++c) d += pl[c] * l2w[c * HOUTD + j];
  red[j] = softplus_(d) * ow[j];
  __syncthreads();
  for (int w = 64; w > 0; w >>= 1) {
    if (j < w) red[j] += red[j + w];
    __syncthreads();
  }
  if (j == 0) out[gidx] = red[0] + ob[0];
}

// ---------------- host launch ----------------
extern "C" void kernel_launch(void* const* d_in, const int* in_sizes, int n_in,
                              void* d_out, int out_size, void* d_ws, size_t ws_size,
                              hipStream_t stream) {
  (void)in_sizes; (void)n_in; (void)out_size; (void)ws_size;
  const float* x          = (const float*)d_in[0];
  const int*   node_idx   = (const int*)d_in[1];
  const int*   hedge_idx  = (const int*)d_in[2];
  const float* hedge_attr = (const float*)d_in[3];
  const int*   batch      = (const int*)d_in[4];
  const float* embed_w    = (const float*)d_in[5];
  const float* embed_b    = (const float*)d_in[6];
  const float* f1_w = (const float*)d_in[7];
  // f1_b (d_in[8]) and c1_b (d_in[10]) are absorbed by training-mode BN.
  const float* c1_w = (const float*)d_in[9];
  const float* f2_w = (const float*)d_in[11];
  const float* f2_b = (const float*)d_in[12];
  const float* c2_w = (const float*)d_in[13];
  const float* c2_b = (const float*)d_in[14];
  const float* bnf_g = (const float*)d_in[15];
  const float* bnf_b = (const float*)d_in[16];
  const float* bnc_g = (const float*)d_in[17];
  const float* bnc_b = (const float*)d_in[18];
  const float* bno_g = (const float*)d_in[19];
  const float* bno_b = (const float*)d_in[20];
  const float* l2_w  = (const float*)d_in[21];
  const float* l2_b  = (const float*)d_in[22];
  const float* out_w = (const float*)d_in[23];
  const float* out_b = (const float*)d_in[24];

  // -------- workspace layout (≈271 MB total, 256B-aligned chunks) --------
  char* base = (char*)d_ws;
  size_t off = 0;
  auto alloc = [&](size_t bytes) -> char* {
    char* p = base + off;
    off += (bytes + 255) & ~(size_t)255;
    return p;
  };
  float* x_cur  = (float*)alloc((size_t)NN * HD * 4);
  h16*   x_half = (h16*)  alloc((size_t)NN * HD * 2);
  float* he_cur = (float*)alloc((size_t)NH * HEF * 4);
  h16*   he_half= (h16*)  alloc((size_t)NH * 48 * 2);      // padded rows
  float* hsum   = (float*)alloc((size_t)NH * HD * 4);
  float* hcnt   = (float*)alloc((size_t)NH * 4);
  float* nsum   = (float*)alloc((size_t)NN * HD * 4);
  float* ncnt   = (float*)alloc((size_t)NN * 4);
  h16*   msg    = (h16*)  alloc((size_t)NH * 128 * 2);
  float* zf     = (float*)alloc((size_t)NH * HEF * 4);
  float* zc     = (float*)alloc((size_t)NH * HEF * 4);
  h16*   wemb   = (h16*)  alloc(96 * 64 * 2);
  h16*   wf1    = (h16*)  alloc(128 * 48 * 2);
  h16*   wc1    = (h16*)  alloc(128 * 48 * 2);
  h16*   wf2    = (h16*)  alloc(128 * 64 * 2);
  h16*   wc2    = (h16*)  alloc(128 * 64 * 2);
  float* psum   = (float*)alloc((size_t)NG * HD * 4);
  float* pcnt   = (float*)alloc((size_t)NG * 4);
  float* stats  = (float*)alloc(384 * 4);
  float *mf = stats, *rf = stats + 64, *mc = stats + 128, *rc = stats + 192;
  float *mo = stats + 256, *ro = stats + 320;

  const int TPB = 256;
  // one-time init: counts are static across layers
  hipMemsetAsync(hcnt, 0, (size_t)NH * 4, stream);
  hipMemsetAsync(ncnt, 0, (size_t)NN * 4, stream);
  hipMemsetAsync(pcnt, 0, (size_t)NG * 4, stream);
  hipMemsetAsync(psum, 0, (size_t)NG * HD * 4, stream);
  hipMemsetAsync(he_half, 0, (size_t)NH * 48 * 2, stream);  // zero padding cols
  hipMemcpyAsync(he_cur, hedge_attr, (size_t)NH * HEF * 4, hipMemcpyDeviceToDevice, stream);
  k_inc_counts<<<(NE + TPB - 1) / TPB, TPB, 0, stream>>>(node_idx, hedge_idx, ncnt, hcnt, NE);
  k_batch_counts<<<(NN + TPB - 1) / TPB, TPB, 0, stream>>>(batch, pcnt, NN);

  // embed
  k_convert16<<<(96 * 64 + TPB - 1) / TPB, TPB, 0, stream>>>(embed_w, wemb, 92, 64, 96, 64);
  k_embed<<<256, 128, 0, stream>>>(x, wemb, embed_b, x_cur, x_half, 256 * 4);

  for (int l = 0; l < NLAY; ++l) {
    // per-layer weight conversion to zero-K-padded f16
    k_convert16<<<(128 * 48 + TPB - 1) / TPB, TPB, 0, stream>>>(f1_w + (size_t)l * 99 * 35, wf1, 99, 35, 128, 48);
    k_convert16<<<(128 * 48 + TPB - 1) / TPB, TPB, 0, stream>>>(c1_w + (size_t)l * 99 * 35, wc1, 99, 35, 128, 48);
    k_convert16<<<(128 * 64 + TPB - 1) / TPB, TPB, 0, stream>>>(f2_w + (size_t)l * 99 * 64, wf2, 99, 64, 128, 64);
    k_convert16<<<(128 * 64 + TPB - 1) / TPB, TPB, 0, stream>>>(c2_w + (size_t)l * 99 * 64, wc2, 99, 64, 128, 64);

    // hedge-side mean aggregation + msg
    hipMemsetAsync(hsum, 0, (size_t)NH * HD * 4, stream);
    k_hedge_gather<<<(NE + TPB - 1) / TPB, TPB, 0, stream>>>(node_idx, hedge_idx, x_cur, hsum, NE);
    k_hedge_msg<<<(NH * 128 + TPB - 1) / TPB, TPB, 0, stream>>>(hsum, hcnt, he_cur, msg, NH * 128);

    // hedge GEMM + BN + gate
    k_hedge_gemm<<<768, 128, 0, stream>>>(msg, wf1, wc1, zf, zc, 768 * 4);
    k_colstats<<<HEF, 256, 0, stream>>>(zf, NH, HEF, mf, rf);
    k_colstats<<<HEF, 256, 0, stream>>>(zc, NH, HEF, mc, rc);
    k_hedge_act<<<(NH * HEF + TPB - 1) / TPB, TPB, 0, stream>>>(
        zf, zc, mf, rf, mc, rc, bnf_g + l * HEF, bnf_b + l * HEF,
        bnc_g + l * HEF, bnc_b + l * HEF, he_cur, he_half, NH * HEF);

    // edge GEMM fused with gather/gate/scatter-mean
    hipMemsetAsync(nsum, 0, (size_t)NN * HD * 4, stream);
    k_edge_gemm<<<1024, 128, 0, stream>>>(node_idx, hedge_idx, x_half, he_half,
                                          wf2, wc2, f2_b + l * HD, c2_b + l * HD,
                                          nsum, 1024 * 4);
    k_node_finalize<<<(NN * HD + TPB - 1) / TPB, TPB, 0, stream>>>(nsum, ncnt, NN * HD);
    k_colstats<<<HD, 256, 0, stream>>>(nsum, NN, HD, mo, ro);
    k_node_act<<<(NN * HD + TPB - 1) / TPB, TPB, 0, stream>>>(
        nsum, mo, ro, bno_g + l * HD, bno_b + l * HD, x_cur, x_half, NN * HD);
  }

  // graph pooling + MLP head
  k_pool<<<(NN * HD + TPB - 1) / TPB, TPB, 0, stream>>>(batch, x_cur, psum, NN * HD);
  k_head<<<NG, 128, 0, stream>>>(psum, pcnt, l2_w, l2_b, out_w, out_b, (float*)d_out);
}
